// GateLinearAttentionNoSilu_14989435863498
// MI455X (gfx1250) — matrix operand order repaired and checked by reading, not compiled, not run on hardware
//
#include <hip/hip_runtime.h>
#include <math.h>

// ---------------------------------------------------------------------------
// B=8, DIM=512, HEADS=8, HEAD_DIM=64, H=W=64 -> N=4096, SCALE=0.125
// ---------------------------------------------------------------------------
#define BB     8
#define DIMC   512
#define NHEAD  8
#define HDIM   64
#define NPIX   4096
#define SCALEF 0.125f

typedef __attribute__((ext_vector_type(16))) __bf16 v16bf;
typedef __attribute__((ext_vector_type(8)))  __bf16 v8bf;
typedef __attribute__((ext_vector_type(8)))  float  v8f;

union Frag { v16bf v; v8bf h[2]; };

// ---------------------------------------------------------------------------
// Bulk f32 -> bf16 conversion (counts are multiples of 8)
// ---------------------------------------------------------------------------
__global__ void cvt_kernel(const float* __restrict__ src, __bf16* __restrict__ dst,
                           int count) {
    int i = (blockIdx.x * blockDim.x + threadIdx.x) * 8;
    if (i + 8 <= count) {
        float f[8];
        __builtin_memcpy(f, src + i, 32);
        __bf16 o[8];
        #pragma unroll
        for (int j = 0; j < 8; ++j) o[j] = (__bf16)f[j];
        __builtin_memcpy(dst + i, o, 16);
    }
}

// ---------------------------------------------------------------------------
// sin/cos (N,64) -> transposed (64,N) so rope passes read coalesced
// ---------------------------------------------------------------------------
__global__ void transpose_sc(const float* __restrict__ sinp, const float* __restrict__ cosp,
                             float* __restrict__ sinT, float* __restrict__ cosT) {
    const int d = blockIdx.x;
    for (int t = threadIdx.x; t < NPIX; t += 256) {
        sinT[(size_t)d * NPIX + t] = sinp[(size_t)t * HDIM + d];
        cosT[(size_t)d * NPIX + t] = cosp[(size_t)t * HDIM + d];
    }
}

// ---------------------------------------------------------------------------
// GEMM: Out[b,o,n] = sum_c W[o,c]*X[b,c,n] + bias[o]
// grid=(N/64, O/128, B), block=256 (8 waves). Ping-pong LDS staging,
// register-prefetched global loads, one barrier per K-step, bf16 WMMA.
// ---------------------------------------------------------------------------
#define TM 128
#define TN 64
#define TK 32
#define SPITCH 40   // bf16 pitch (80B) keeps 16B-aligned fragment chunks

__global__ void gemm_wmma(const __bf16* __restrict__ Wbf, const float* __restrict__ X,
                          const float* __restrict__ bias, float* __restrict__ Out,
                          int O, int C, int N) {
    __shared__ __bf16 sA[2][TM * SPITCH];
    __shared__ __bf16 sB[2][TN * SPITCH];
    const int tid  = threadIdx.x;
    const int lane = tid & 31;
    const int wave = tid >> 5;
    const int mrow = lane & 15;
    const int hi   = (lane >> 4) & 1;
    const int n0   = blockIdx.x * TN;
    const int o0b  = blockIdx.y * TM;
    const float* Xb = X + (size_t)blockIdx.z * C * N;
    float*       Ob = Out + (size_t)blockIdx.z * O * N;

    const int arow  = tid >> 1;   // 0..127
    const int ahalf = tid & 1;    // 16 bf16 each
    const int bc    = tid >> 3;   // 0..31 (c)
    const int bpart = tid & 7;    // 8 n each

    const __bf16* wsrc = Wbf + (size_t)(o0b + arow) * C + ahalf * 16;
    const float*  xsrc = Xb + (size_t)bc * N + n0 + bpart * 8;

    __bf16 ra[16];
    float  rb[8];
    // prologue: prefetch tile kb=0 into registers
    __builtin_memcpy(ra, wsrc, 32);
    __builtin_memcpy(rb, xsrc, 32);

    v8f acc[4] = {};
    int cur = 0;
    for (int kb = 0; kb < C; kb += TK, cur ^= 1) {
        // commit staged registers to LDS buffer `cur`
        __builtin_memcpy(&sA[cur][arow * SPITCH + ahalf * 16], ra, 32);
        #pragma unroll
        for (int i = 0; i < 8; ++i)
            sB[cur][(bpart * 8 + i) * SPITCH + bc] = (__bf16)rb[i];
        __syncthreads();
        // issue next tile's global loads (in flight under this tile's math)
        if (kb + TK < C) {
            __builtin_memcpy(ra, wsrc + kb + TK, 32);
            __builtin_memcpy(rb, xsrc + (size_t)(kb + TK) * N, 32);
        }
        // fragments: batch all LDS loads, then burst the 4 WMMAs
        Frag a;
        const __bf16* ap = &sA[cur][(wave * 16 + mrow) * SPITCH];
        __builtin_memcpy(&a.h[0], ap + hi * 16, 16);
        __builtin_memcpy(&a.h[1], ap + hi * 16 + 8, 16);
        Frag b4[4];
        #pragma unroll
        for (int t = 0; t < 4; ++t) {
            const __bf16* bp = &sB[cur][(t * 16 + mrow) * SPITCH + hi * 16];
            __builtin_memcpy(&b4[t].h[0], bp, 16);
            __builtin_memcpy(&b4[t].h[1], bp + 8, 16);
        }
        #pragma unroll
        for (int t = 0; t < 4; ++t)
            acc[t] = __builtin_amdgcn_wmma_f32_16x16x32_bf16(
                false, a.v, false, b4[t].v, (short)0, acc[t], false, false);
    }
    #pragma unroll
    for (int t = 0; t < 4; ++t) {
        const int n = n0 + t * 16 + mrow;
        #pragma unroll
        for (int r = 0; r < 8; ++r) {
            const int o = o0b + wave * 16 + r + hi * 8;
            Ob[(size_t)o * N + n] = acc[t][r] + bias[o];
        }
    }
}

// ---------------------------------------------------------------------------
// Block reductions
// ---------------------------------------------------------------------------
__device__ __forceinline__ float blockSum(float v, float* sred) {
    const int tid = threadIdx.x;
    sred[tid] = v; __syncthreads();
    for (int s = 128; s > 0; s >>= 1) {
        if (tid < s) sred[tid] += sred[tid + s];
        __syncthreads();
    }
    float r = sred[0]; __syncthreads(); return r;
}
__device__ __forceinline__ float blockMax(float v, float* sred) {
    const int tid = threadIdx.x;
    sred[tid] = v; __syncthreads();
    for (int s = 128; s > 0; s >>= 1) {
        if (tid < s) sred[tid] = fmaxf(sred[tid], sred[tid + s]);
        __syncthreads();
    }
    float r = sred[0]; __syncthreads(); return r;
}

// ---------------------------------------------------------------------------
// Per-(b,h) stats. q <- elu(q)+1 in place; softmax gating; emits
//   krope[bh][d][t] bf16, qrope[bh][t][d] bf16, zbuf[bh][t]
// grid=64, block=256. Logits in 16 KB of LDS.
// ---------------------------------------------------------------------------
__global__ void stats_kernel(float* __restrict__ qkvo,
                             const float* __restrict__ sinp, const float* __restrict__ cosp,
                             const float* __restrict__ sinT, const float* __restrict__ cosT,
                             float* __restrict__ zbuf,
                             __bf16* __restrict__ krope, __bf16* __restrict__ qrope) {
    __shared__ float sEff[NPIX];
    __shared__ float sRed[256];
    __shared__ float sQm[HDIM];
    __shared__ float sKm[HDIM];
    const int tid = threadIdx.x;
    const int bh = blockIdx.x;
    const int b = bh >> 3, h = bh & 7;
    float* q = qkvo + ((size_t)b * 2048 + h * HDIM) * NPIX;
    float* k = qkvo + ((size_t)b * 2048 + DIMC + h * HDIM) * NPIX;

    // Pass A: q <- elu(q)+1, q_mean
    for (int d = 0; d < HDIM; ++d) {
        float loc = 0.f;
        for (int t = tid; t < NPIX; t += 256) {
            float v = q[(size_t)d * NPIX + t];
            v = (v > 0.f) ? (v + 1.f) : __expf(v);
            q[(size_t)d * NPIX + t] = v;
            loc += v;
        }
        float s = blockSum(loc, sRed);
        if (tid == 0) sQm[d] = s * (1.f / NPIX);
        __syncthreads();
    }

    // Pass B: k <- elu(k)+1, logits = q_mean . k[t]
    for (int t = tid; t < NPIX; t += 256) sEff[t] = 0.f;
    __syncthreads();
    for (int d = 0; d < HDIM; ++d) {
        const float qm = sQm[d];
        for (int t = tid; t < NPIX; t += 256) {
            float v = k[(size_t)d * NPIX + t];
            v = (v > 0.f) ? (v + 1.f) : __expf(v);
            k[(size_t)d * NPIX + t] = v;
            sEff[t] += qm * v;
        }
    }
    __syncthreads();

    // softmax over t
    float lm = -3.402823e38f;
    for (int t = tid; t < NPIX; t += 256) lm = fmaxf(lm, SCALEF * sEff[t]);
    const float gm = blockMax(lm, sRed);
    float ls = 0.f;
    for (int t = tid; t < NPIX; t += 256) {
        float e = __expf(SCALEF * sEff[t] - gm);
        sEff[t] = e; ls += e;
    }
    const float gs = blockSum(ls, sRed);
    const float inv = 1.f / gs;
    for (int t = tid; t < NPIX; t += 256) sEff[t] *= inv;
    __syncthreads();

    // Pass C: scale + rope k (pairs), emit bf16 krope, k_mean
    __bf16* krb = krope + (size_t)bh * HDIM * NPIX;
    for (int d0 = 0; d0 < HDIM; d0 += 2) {
        const size_t r0 = (size_t)d0 * NPIX, r1 = r0 + NPIX;
        const float* st0 = sinT + r0; const float* st1 = sinT + r1;
        const float* ct0 = cosT + r0; const float* ct1 = cosT + r1;
        float s0 = 0.f, s1 = 0.f;
        for (int t = tid; t < NPIX; t += 256) {
            const float e = sEff[t] * (float)NPIX;
            const float v0 = k[r0 + t] * e;
            const float v1 = k[r1 + t] * e;
            s0 += v0; s1 += v1;
            krb[r0 + t] = (__bf16)(v0 * ct0[t] - v1 * st0[t]);
            krb[r1 + t] = (__bf16)(v1 * ct1[t] + v0 * st1[t]);
        }
        const float S0 = blockSum(s0, sRed);
        const float S1 = blockSum(s1, sRed);
        if (tid == 0) { sKm[d0] = S0 * (1.f / NPIX); sKm[d0 + 1] = S1 * (1.f / NPIX); }
        __syncthreads();
    }

    // Pass D: z and bf16 qrope [t][d]
    for (int t = tid; t < NPIX; t += 256) {
        float acc = 0.f;
        const float* srow = sinp + (size_t)t * HDIM;
        const float* crow = cosp + (size_t)t * HDIM;
        __bf16* qr = qrope + ((size_t)bh * NPIX + t) * HDIM;
        #pragma unroll 4
        for (int d0 = 0; d0 < HDIM; d0 += 2) {
            const float q0 = q[(size_t)d0 * NPIX + t];
            const float q1 = q[(size_t)(d0 + 1) * NPIX + t];
            acc += q0 * sKm[d0] + q1 * sKm[d0 + 1];
            qr[d0]     = (__bf16)(q0 * crow[d0] - q1 * srow[d0]);
            qr[d0 + 1] = (__bf16)(q1 * crow[d0 + 1] + q0 * srow[d0 + 1]);
        }
        zbuf[(size_t)bh * NPIX + t] = 1.f / (acc + 1e-6f);
    }
}

// ---------------------------------------------------------------------------
// kv[bh][d][e] = (1/N) * sum_t krope[t][d] * v[t][e]
// grid=64, block=512 (16 waves -> 16 output tiles). Pure b128 loads + WMMA,
// 2 K-steps of loads batched ahead of the accumulator chain.
// ---------------------------------------------------------------------------
__global__ void kv_kernel(const __bf16* __restrict__ krope, const __bf16* __restrict__ vbf,
                          float* __restrict__ kvbuf) {
    const int bh = blockIdx.x;
    const int lane = threadIdx.x & 31;
    const int wave = threadIdx.x >> 5;
    const int tileD = wave >> 2, tileE = wave & 3;
    const int mrow = lane & 15;
    const int hi = (lane >> 4) & 1;

    const __bf16* kr = krope + ((size_t)bh * HDIM + tileD * 16 + mrow) * NPIX;
    const __bf16* vr = vbf   + ((size_t)bh * HDIM + tileE * 16 + mrow) * NPIX;

    v8f acc = {};
    for (int tb = 0; tb < NPIX; tb += 64) {
        Frag a0, b0, a1, b1;
        __builtin_memcpy(&a0.h[0], kr + tb + hi * 16, 16);
        __builtin_memcpy(&a0.h[1], kr + tb + hi * 16 + 8, 16);
        __builtin_memcpy(&b0.h[0], vr + tb + hi * 16, 16);
        __builtin_memcpy(&b0.h[1], vr + tb + hi * 16 + 8, 16);
        __builtin_memcpy(&a1.h[0], kr + tb + 32 + hi * 16, 16);
        __builtin_memcpy(&a1.h[1], kr + tb + 32 + hi * 16 + 8, 16);
        __builtin_memcpy(&b1.h[0], vr + tb + 32 + hi * 16, 16);
        __builtin_memcpy(&b1.h[1], vr + tb + 32 + hi * 16 + 8, 16);
        acc = __builtin_amdgcn_wmma_f32_16x16x32_bf16(
            false, a0.v, false, b0.v, (short)0, acc, false, false);
        acc = __builtin_amdgcn_wmma_f32_16x16x32_bf16(
            false, a1.v, false, b1.v, (short)0, acc, false, false);
    }
    #pragma unroll
    for (int r = 0; r < 8; ++r) {
        const int dR = tileD * 16 + r + hi * 8;
        const int eC = tileE * 16 + mrow;
        kvbuf[(size_t)bh * (HDIM * HDIM) + dR * HDIM + eC] = acc[r] * (1.f / NPIX);
    }
}

// ---------------------------------------------------------------------------
// res[b, h*64+e, t] = z[t] * sum_d qrope[t][d] * kv[d][e]
// grid=(32,64), block=256 (8 waves). kv transposed into LDS as bf16.
// ---------------------------------------------------------------------------
#define KVP 72   // bf16 pitch for sKVT rows (144B, 16B-aligned chunks)

__global__ void res_kernel(const __bf16* __restrict__ qrope, const float* __restrict__ kvbuf,
                           const float* __restrict__ zbuf, float* __restrict__ resbuf) {
    __shared__ __bf16 sKVT[HDIM * KVP];   // [e][d]
    const int tid = threadIdx.x;
    const int bh = blockIdx.y;
    const int b = bh >> 3, h = bh & 7;
    for (int i = tid; i < HDIM * HDIM; i += 256) {
        const int d = i >> 6, e = i & 63;
        sKVT[e * KVP + d] = (__bf16)kvbuf[(size_t)bh * (HDIM * HDIM) + i];
    }
    __syncthreads();

    const int lane = tid & 31;
    const int wave = tid >> 5;
    const int mrow = lane & 15;
    const int hi = (lane >> 4) & 1;
    const int tRow = blockIdx.x * 128 + wave * 16 + mrow;
    const __bf16* qr = qrope + ((size_t)bh * NPIX + tRow) * HDIM;

    v8f acc[4] = {};
    #pragma unroll
    for (int kb = 0; kb < HDIM; kb += 32) {
        Frag a;
        __builtin_memcpy(&a.h[0], qr + kb + hi * 16, 16);
        __builtin_memcpy(&a.h[1], qr + kb + hi * 16 + 8, 16);
        Frag b4[4];
        #pragma unroll
        for (int te = 0; te < 4; ++te) {
            const __bf16* bp = &sKVT[(te * 16 + mrow) * KVP + kb + hi * 16];
            __builtin_memcpy(&b4[te].h[0], bp, 16);
            __builtin_memcpy(&b4[te].h[1], bp + 8, 16);
        }
        #pragma unroll
        for (int te = 0; te < 4; ++te)
            acc[te] = __builtin_amdgcn_wmma_f32_16x16x32_bf16(
                false, a.v, false, b4[te].v, (short)0, acc[te], false, false);
    }
    #pragma unroll
    for (int te = 0; te < 4; ++te) {
        const int ec = te * 16 + mrow;
        float* dst = resbuf + ((size_t)b * DIMC + h * HDIM + ec) * NPIX;
        #pragma unroll
        for (int r = 0; r < 8; ++r) {
            const int tt = blockIdx.x * 128 + wave * 16 + r + hi * 8;
            dst[tt] = acc[te][r] * zbuf[(size_t)bh * NPIX + tt];
        }
    }
}

// ---------------------------------------------------------------------------
// resbuf <- (resbuf + depthwise5x5(v) + b_lepe) * o_gate   (in place)
// ---------------------------------------------------------------------------
__global__ void lepe_gate_kernel(const float* __restrict__ qkvo,
                                 const float* __restrict__ wlepe,
                                 const float* __restrict__ blepe,
                                 float* __restrict__ resbuf) {
    __shared__ float wf[25];
    const int c = blockIdx.x & (DIMC - 1);
    const int b = blockIdx.x >> 9;
    if (threadIdx.x < 25) wf[threadIdx.x] = wlepe[(size_t)c * 25 + threadIdx.x];
    __syncthreads();
    const float* v  = qkvo + ((size_t)b * 2048 + 2 * DIMC + c) * NPIX;
    const float* og = qkvo + ((size_t)b * 2048 + 3 * DIMC + c) * NPIX;
    float* r = resbuf + ((size_t)b * DIMC + c) * NPIX;
    const float bl = blepe[c];
    for (int n = threadIdx.x; n < NPIX; n += 256) {
        const int row = n >> 6, col = n & 63;
        float s = bl;
        #pragma unroll
        for (int i = 0; i < 5; ++i) {
            const int rr = row + i - 2;
            if (rr < 0 || rr > 63) continue;
            #pragma unroll
            for (int j = 0; j < 5; ++j) {
                const int cc = col + j - 2;
                if (cc < 0 || cc > 63) continue;
                s += wf[i * 5 + j] * v[rr * 64 + cc];
            }
        }
        r[n] = (r[n] + s) * og[n];
    }
}

// ---------------------------------------------------------------------------
// Launch. Inputs: x, sin, cos, w_qkvo, b_qkvo, w_lepe, b_lepe, w_proj, b_proj
// ---------------------------------------------------------------------------
extern "C" void kernel_launch(void* const* d_in, const int* in_sizes, int n_in,
                              void* d_out, int out_size, void* d_ws, size_t ws_size,
                              hipStream_t stream) {
    (void)in_sizes; (void)n_in; (void)out_size; (void)ws_size;
    const float* x      = (const float*)d_in[0];
    const float* sinp   = (const float*)d_in[1];
    const float* cosp   = (const float*)d_in[2];
    const float* w_qkvo = (const float*)d_in[3];
    const float* b_qkvo = (const float*)d_in[4];
    const float* w_lepe = (const float*)d_in[5];
    const float* b_lepe = (const float*)d_in[6];
    const float* w_proj = (const float*)d_in[7];
    const float* b_proj = (const float*)d_in[8];
    float* out = (float*)d_out;

    // f32 scratch
    float* qkvo   = (float*)d_ws;                              // 67,108,864
    float* resbuf = qkvo   + (size_t)BB * 4 * DIMC * NPIX;     // 16,777,216
    float* kvbuf  = resbuf + (size_t)BB * DIMC * NPIX;         //    262,144
    float* zbuf   = kvbuf  + (size_t)BB * NHEAD * HDIM * HDIM; //    262,144
    float* sinT   = zbuf   + (size_t)BB * NHEAD * NPIX;        //    262,144
    float* cosT   = sinT   + (size_t)NPIX * HDIM;              //    262,144
    // bf16 scratch
    __bf16* wqkvo_bf = (__bf16*)(cosT + (size_t)NPIX * HDIM);
    __bf16* wproj_bf = wqkvo_bf + (size_t)2048 * DIMC;
    __bf16* vbf      = wproj_bf + (size_t)DIMC * DIMC;
    __bf16* krope    = vbf      + (size_t)BB * DIMC * NPIX;
    __bf16* qrope    = krope    + (size_t)BB * DIMC * NPIX;

    const dim3 blk(256);
    cvt_kernel<<<dim3((2048 * DIMC) / 2048), blk, 0, stream>>>(w_qkvo, wqkvo_bf, 2048 * DIMC);
    cvt_kernel<<<dim3((DIMC * DIMC) / 2048), blk, 0, stream>>>(w_proj, wproj_bf, DIMC * DIMC);
    transpose_sc<<<dim3(HDIM), blk, 0, stream>>>(sinp, cosp, sinT, cosT);

    // 1) qkvo = w_qkvo @ x + b_qkvo
    gemm_wmma<<<dim3(NPIX / TN, 2048 / TM, BB), blk, 0, stream>>>(
        wqkvo_bf, x, b_qkvo, qkvo, 2048, DIMC, NPIX);
    // v channels -> bf16 [bh][e][t]
    for (int b = 0; b < BB; ++b)
        cvt_kernel<<<dim3((DIMC * NPIX) / 2048), blk, 0, stream>>>(
            qkvo + ((size_t)b * 2048 + 2 * DIMC) * NPIX,
            vbf + (size_t)b * DIMC * NPIX, DIMC * NPIX);
    // 2) stats: elu, softmax gating, rope -> krope/qrope/z
    stats_kernel<<<dim3(BB * NHEAD), blk, 0, stream>>>(
        qkvo, sinp, cosp, sinT, cosT, zbuf, krope, qrope);
    // 3) kv = krope^T @ v / N
    kv_kernel<<<dim3(BB * NHEAD), dim3(512), 0, stream>>>(krope, vbf, kvbuf);
    // 4) res = qrope @ kv * z
    res_kernel<<<dim3(NPIX / 128, BB * NHEAD), blk, 0, stream>>>(
        qrope, kvbuf, zbuf, resbuf);
    // 5) resbuf = (resbuf + lepe(v)) * o_gate
    lepe_gate_kernel<<<dim3(BB * DIMC), blk, 0, stream>>>(qkvo, w_lepe, b_lepe, resbuf);
    // 6) out = w_proj @ resbuf + b_proj
    gemm_wmma<<<dim3(NPIX / TN, DIMC / TM, BB), blk, 0, stream>>>(
        wproj_bf, resbuf, b_proj, out, DIMC, DIMC, NPIX);
}
